// CheapChannelV1_4784593568211
// MI455X (gfx1250) — matrix operand order repaired and checked by reading, not compiled, not run on hardware
//
#include <hip/hip_runtime.h>
#include <math.h>

typedef __attribute__((ext_vector_type(16))) _Float16 v16h;
typedef __attribute__((ext_vector_type(8)))  float    v8f;

union HV16 { v16h v; uint4 q[2]; };
union PK8  { uint4 q; _Float16 h[8]; };

__device__ __forceinline__ v8f wmma16(v16h a, v16h b, v8f c) {
    // D = A(16x32) x B(32x16) + C, f32 accumulate
    return __builtin_amdgcn_wmma_f32_16x16x32_f16(
        /*neg_a=*/false, a, /*neg_b=*/false, b,
        /*c_mod=*/(short)0, c, /*reuse_a=*/false, /*reuse_b=*/false);
}

// ---------------------------------------------------------------------------
// Prep: fold channel-shuffle permutation into f16 weight matrices.
// W'[o][i] = W[o][(i % (C/8))*8 + i/(C/8)]
// ---------------------------------------------------------------------------
__global__ void prep_weights(const float* __restrict__ w1,
                             const float* __restrict__ w2,
                             const float* __restrict__ w3,
                             _Float16* __restrict__ W1,
                             _Float16* __restrict__ W2,
                             _Float16* __restrict__ W3) {
    int gid = blockIdx.x * blockDim.x + threadIdx.x;
    if (gid < 4096) {                       // 64x64, C8=8
        int o = gid / 64, i = gid % 64;
        W1[gid] = (_Float16)w1[o * 64 + (i % 8) * 8 + i / 8];
    } else if (gid < 13312) {               // 96x96, C8=12
        int g = gid - 4096;
        int o = g / 96, i = g % 96;
        W2[g] = (_Float16)w2[o * 96 + (i % 12) * 8 + i / 12];
    } else if (gid < 29696) {               // 128x128, C8=16
        int g = gid - 13312;
        int o = g / 128, i = g % 128;
        W3[g] = (_Float16)w3[o * 128 + (i % 16) * 8 + i / 16];
    }
}

// ---------------------------------------------------------------------------
// k x k max pool of one 32-channel group of x (NCHW, C=128, 256x256)
// ---------------------------------------------------------------------------
__global__ void pool_kernel(const float* __restrict__ x, float* __restrict__ P,
                            int cg, int k, int oH, int oW, int total) {
    int gid = blockIdx.x * blockDim.x + threadIdx.x;
    if (gid >= total) return;
    int ow = gid % oW;
    int oh = (gid / oW) % oH;
    int c  = (gid / (oW * oH)) % 32;
    int b  = gid / (oW * oH * 32);
    const float* src = x + (((long)b * 128 + cg + c) * 256 + (long)oh * k) * 256 + (long)ow * k;
    float m = -3.402823466e38f;
    for (int i = 0; i < k; ++i)
        for (int j = 0; j < k; ++j)
            m = fmaxf(m, src[i * 256 + j]);
    P[gid] = m;
}

// ---------------------------------------------------------------------------
// Depthwise 3x3 SAME conv (cross-correlation) + bias, f32 in -> f16 out.
// ---------------------------------------------------------------------------
__global__ void dw_kernel(const float* __restrict__ in, long batchStride, long chanStride,
                          int H, int W,
                          const float* __restrict__ wdw, const float* __restrict__ bdw,
                          _Float16* __restrict__ out, int total) {
    int gid = blockIdx.x * blockDim.x + threadIdx.x;
    if (gid >= total) return;
    int w = gid % W;
    int h = (gid / W) % H;
    int c = (gid / (W * H)) % 32;
    int b = gid / (W * H * 32);
    const float* src = in + (long)b * batchStride + (long)c * chanStride;
    const float* wc  = wdw + c * 9;
    float s = bdw[c];
    #pragma unroll
    for (int kh = -1; kh <= 1; ++kh) {
        int hh = h + kh;
        if (hh < 0 || hh >= H) continue;
        #pragma unroll
        for (int kw = -1; kw <= 1; ++kw) {
            int ww = w + kw;
            if (ww < 0 || ww >= W) continue;
            s += src[(long)hh * W + ww] * wc[(kh + 1) * 3 + (kw + 1)];
        }
    }
    out[gid] = (_Float16)s;
}

// ---------------------------------------------------------------------------
// One GEMM stage over a 16-pixel tile held in LDS actT[pixel][channel].
//   A = W' (MTx16 out-ch  x  K in-ch, f16 row-major), B = actT columns.
// Non-final: writes f16 result back into actT[:, 0:MT*16].
// Final: exact GELU * x -> out.
// ---------------------------------------------------------------------------
template <int MT, int KC, bool FINAL>
__device__ __forceinline__ void gemm_stage(const _Float16* __restrict__ Wp,
                                           const float* __restrict__ bias,
                                           _Float16 (*actT)[136], int lane,
                                           const float* __restrict__ xin,
                                           float* __restrict__ outp, long pixbase) {
    const int p  = lane & 15;
    const int hb = lane >> 4;
    const int K  = KC * 32;

    // B fragments: element e of lane = actT[p][kc*32 + 16*hb + e]
    v16h bf[KC];
    #pragma unroll
    for (int kc = 0; kc < KC; ++kc) {
        const uint4* s = (const uint4*)&actT[p][kc * 32 + hb * 16];
        HV16 u;
        u.q[0] = s[0];
        u.q[1] = s[1];
        bf[kc] = u.v;
    }

    v8f acc[MT];
    #pragma unroll
    for (int mt = 0; mt < MT; ++mt) {
        #pragma unroll
        for (int r = 0; r < 8; ++r)
            acc[mt][r] = bias[mt * 16 + hb * 8 + r];
        const _Float16* wrow = Wp + (long)(mt * 16 + p) * K;
        #pragma unroll
        for (int kc = 0; kc < KC; ++kc) {
            // A fragment: elem e -> K = kc*32 + 16*hb + e (matches B)
            HV16 ua;
            ua.q[0] = *(const uint4*)(wrow + kc * 32 + hb * 16);
            ua.q[1] = *(const uint4*)(wrow + kc * 32 + hb * 16 + 8);
            acc[mt] = wmma16(ua.v, bf[kc], acc[mt]);
        }
    }

    __syncthreads();
    if (!FINAL) {
        // D layout: VGPR r holds (M = mt*16 + 8*hb + r, N = p) -> one b128 store
        #pragma unroll
        for (int mt = 0; mt < MT; ++mt) {
            PK8 pk;
            #pragma unroll
            for (int r = 0; r < 8; ++r) pk.h[r] = (_Float16)acc[mt][r];
            *(uint4*)&actT[p][mt * 16 + hb * 8] = pk.q;
        }
        __syncthreads();
    } else {
        #pragma unroll
        for (int mt = 0; mt < MT; ++mt) {
            #pragma unroll
            for (int r = 0; r < 8; ++r) {
                int  M   = mt * 16 + hb * 8 + r;
                long idx = pixbase + (long)M * 65536 + p;
                float v  = acc[mt][r];
                float g  = 0.5f * v * (1.0f + erff(v * 0.70710678118654752f));
                outp[idx] = g * xin[idx];
            }
        }
    }
}

// ---------------------------------------------------------------------------
// Fused pointwise chain: one wave32 per 16-pixel row segment.
// ---------------------------------------------------------------------------
__global__ __launch_bounds__(32) void fused_pw_kernel(
    const _Float16* __restrict__ S0, const _Float16* __restrict__ T1,
    const _Float16* __restrict__ T2, const _Float16* __restrict__ T3,
    const _Float16* __restrict__ W1, const _Float16* __restrict__ W2,
    const _Float16* __restrict__ W3,
    const float* __restrict__ b1, const float* __restrict__ b2,
    const float* __restrict__ b3,
    const float* __restrict__ x, float* __restrict__ out) {
    __shared__ __align__(16) _Float16 actT[16][136];  // [pixel][channel], padded stride

    int tile = blockIdx.x;        // 32768 tiles
    int b    = tile >> 12;        // 4096 tiles per batch image
    int t    = tile & 4095;
    int h    = t >> 4;            // 16 tiles per row
    int w0   = (t & 15) << 4;

    int lane = threadIdx.x & 31;
    int p    = lane & 15;
    int hb   = lane >> 4;
    int w    = w0 + p;

    // Stage all 128 source channels for this tile:
    //   [0:32)=s0, [32:64)=up2(t1), [64:96)=up4(t2), [96:128)=up8(t3)
    #pragma unroll 4
    for (int cc = 0; cc < 64; ++cc) {
        int c = hb * 64 + cc;
        _Float16 v;
        if (c < 32)
            v = S0[(((long)b * 32 + c) * 256 + h) * 256 + w];
        else if (c < 64)
            v = T1[(((long)b * 32 + (c - 32)) * 128 + (h >> 1)) * 128 + (w >> 1)];
        else if (c < 96)
            v = T2[(((long)b * 32 + (c - 64)) * 64 + (h >> 2)) * 64 + (w >> 2)];
        else
            v = T3[(((long)b * 32 + (c - 96)) * 32 + (h >> 3)) * 32 + (w >> 3)];
        actT[p][c] = v;
    }
    __syncthreads();

    long pixbase = (long)b * 128 * 65536 + (long)h * 256 + w0;

    gemm_stage<4, 2, false>(W1, b1, actT, lane, nullptr, nullptr, 0);  // 64 -> 64
    gemm_stage<6, 3, false>(W2, b2, actT, lane, nullptr, nullptr, 0);  // 96 -> 96
    gemm_stage<8, 4, true >(W3, b3, actT, lane, x, out, pixbase);      // 128 -> 128
}

// ---------------------------------------------------------------------------
// Workspace layout (bytes)
// ---------------------------------------------------------------------------
static const size_t OFF_S0 = 0;               // f16  8*32*256*256
static const size_t OFF_T1 = 33554432;        // f16  8*32*128*128
static const size_t OFF_T2 = 41943040;        // f16  8*32*64*64
static const size_t OFF_T3 = 44040192;        // f16  8*32*32*32
static const size_t OFF_P1 = 44564480;        // f32  8*32*128*128
static const size_t OFF_P2 = 61341696;        // f32  8*32*64*64
static const size_t OFF_P3 = 65536000;        // f32  8*32*32*32
static const size_t OFF_W1 = 66584576;        // f16  64*64
static const size_t OFF_W2 = 66592768;        // f16  96*96
static const size_t OFF_W3 = 66611200;        // f16  128*128  (end 66,643,968)

extern "C" void kernel_launch(void* const* d_in, const int* in_sizes, int n_in,
                              void* d_out, int out_size, void* d_ws, size_t ws_size,
                              hipStream_t stream) {
    (void)in_sizes; (void)n_in; (void)out_size; (void)ws_size;
    const float* x   = (const float*)d_in[0];
    const float* wdw = (const float*)d_in[1];   // (4,32,3,3)
    const float* bdw = (const float*)d_in[2];   // (4,32)
    const float* wf1 = (const float*)d_in[3];
    const float* bf1 = (const float*)d_in[4];
    const float* wf2 = (const float*)d_in[5];
    const float* bf2 = (const float*)d_in[6];
    const float* wf3 = (const float*)d_in[7];
    const float* bf3 = (const float*)d_in[8];
    float* out = (float*)d_out;
    char*  ws  = (char*)d_ws;

    _Float16* S0 = (_Float16*)(ws + OFF_S0);
    _Float16* T1 = (_Float16*)(ws + OFF_T1);
    _Float16* T2 = (_Float16*)(ws + OFF_T2);
    _Float16* T3 = (_Float16*)(ws + OFF_T3);
    float*    P1 = (float*)(ws + OFF_P1);
    float*    P2 = (float*)(ws + OFF_P2);
    float*    P3 = (float*)(ws + OFF_P3);
    _Float16* W1 = (_Float16*)(ws + OFF_W1);
    _Float16* W2 = (_Float16*)(ws + OFF_W2);
    _Float16* W3 = (_Float16*)(ws + OFF_W3);

    prep_weights<<<(29696 + 255) / 256, 256, 0, stream>>>(wf1, wf2, wf3, W1, W2, W3);

    pool_kernel<<<4194304 / 256, 256, 0, stream>>>(x, P1, 32, 2, 128, 128, 4194304);
    pool_kernel<<<1048576 / 256, 256, 0, stream>>>(x, P2, 64, 4, 64, 64, 1048576);
    pool_kernel<<< 262144 / 256, 256, 0, stream>>>(x, P3, 96, 8, 32, 32, 262144);

    // level 0: directly on x group 0 (channel stride 65536, batch stride 128*65536)
    dw_kernel<<<16777216 / 256, 256, 0, stream>>>(x,  (long)128 * 65536, 65536, 256, 256,
                                                  wdw + 0 * 288, bdw + 0,  S0, 16777216);
    dw_kernel<<< 4194304 / 256, 256, 0, stream>>>(P1, (long)32 * 16384, 16384, 128, 128,
                                                  wdw + 1 * 288, bdw + 32, T1, 4194304);
    dw_kernel<<< 1048576 / 256, 256, 0, stream>>>(P2, (long)32 * 4096,  4096,  64,  64,
                                                  wdw + 2 * 288, bdw + 64, T2, 1048576);
    dw_kernel<<<  262144 / 256, 256, 0, stream>>>(P3, (long)32 * 1024,  1024,  32,  32,
                                                  wdw + 3 * 288, bdw + 96, T3, 262144);

    fused_pw_kernel<<<32768, 32, 0, stream>>>(S0, T1, T2, T3, W1, W2, W3,
                                              bf1, bf2, bf3, x, out);
}
